// TCN_63694364999851
// MI455X (gfx1250) — matrix operand order repaired and checked by reading, not compiled, not run on hardware
//
#include <hip/hip_runtime.h>
#include <hip/hip_bf16.h>
#include <stdint.h>

// ---------------- constants ----------------
#define EPS_BN 1e-5f
#define B_   64
#define L_   2048
#define H_   256
#define SPAD 192      // static feature dim padded (181 -> 192)
#define SDIM 181
#define DIN  640      // 128 + 256 + 256
#define FCL  256
#define EMB  128

typedef __attribute__((ext_vector_type(16))) _Float16     v16h;
typedef __attribute__((ext_vector_type(8)))  float        v8f;
typedef __attribute__((ext_vector_type(8)))  unsigned int v8u;
typedef __attribute__((ext_vector_type(4)))  int          v4i;

// ---- CDNA5 async global->LDS copy (ASYNCcnt path), guarded for portability ----
#if defined(__HIP_DEVICE_COMPILE__) && __has_builtin(__builtin_amdgcn_global_load_async_to_lds_b128)
#define USE_ASYNC 1
#else
#define USE_ASYNC 0
#endif

#if USE_ASYNC
typedef __attribute__((address_space(1))) v4i gv4i_t;  // global (prints as __device__)
typedef __attribute__((address_space(3))) v4i lv4i_t;  // LDS    (prints as __shared__)
__device__ __forceinline__ void async_cp16(const _Float16* g, _Float16* l) {
  // copies 16 bytes global -> LDS without passing through VGPRs
  __builtin_amdgcn_global_load_async_to_lds_b128((gv4i_t*)g, (lv4i_t*)l, 0, 0);
}
__device__ __forceinline__ void wait_async0() {
#if __has_builtin(__builtin_amdgcn_s_wait_asynccnt)
  __builtin_amdgcn_s_wait_asynccnt(0);
#else
  asm volatile("s_wait_asynccnt 0x0" ::: "memory");
#endif
}
#endif

__device__ __forceinline__ float selu_f(float x) {
  const float sc = 1.0507009873554805f, al = 1.6732632423543772f;
  return sc * (x > 0.f ? x : al * (__expf(x) - 1.f));
}

// ---- WMMA fragment loaders from LDS (f16, 16x16x32), per CDNA5 ISA layouts ----
// Packed loads: each VGPR holds two f16 at consecutive K -> one ds_load_b32 each.

// A 16x32 (MxK), LDS row-major [m][k], ld even (4B-aligned rows).
__device__ __forceinline__ v16h load_a_frag(const _Float16* base, int ld) {
  const int lane = threadIdx.x & 31;
  const int m = lane & 15;
  const int kb = (lane >> 4) << 3;
  const unsigned int* p = (const unsigned int*)(base + m * ld);
  v8u u;
#pragma unroll
  for (int i = 0; i < 8; ++i) {
    int k = ((i < 4) ? 0 : 16) + kb + ((i & 3) << 1);
    u[i] = p[k >> 1];
  }
  union { v8u u; v16h h; } cvt; cvt.u = u;
  return cvt.h;
}

// B 32x16 (KxN), LDS stored TRANSPOSED as [n][k], ld even.
__device__ __forceinline__ v16h load_b_fragT(const _Float16* base, int ld) {
  const int lane = threadIdx.x & 31;
  const int n = lane & 15;
  const int kb = (lane >> 4) << 3;
  const unsigned int* p = (const unsigned int*)(base + n * ld + kb);
  v8u u;
#pragma unroll
  for (int i = 0; i < 8; ++i) u[i] = p[(i < 4) ? i : i + 4];
  union { v8u u; v16h h; } cvt; cvt.u = u;
  return cvt.h;
}

// ---------------- elementwise prep kernels ----------------

__global__ void bn_cast_f32(const float* __restrict__ x, _Float16* __restrict__ y,
                            const float* __restrict__ g, const float* __restrict__ bt,
                            const float* __restrict__ mu, const float* __restrict__ var,
                            int Cin, int Cpad, int useBN) {
  size_t idx = (size_t)blockIdx.x * blockDim.x + threadIdx.x;
  size_t total = (size_t)B_ * Cpad * L_;
  if (idx >= total) return;
  int t = (int)(idx % L_);
  int c = (int)((idx / L_) % Cpad);
  int b = (int)(idx / ((size_t)L_ * Cpad));
  float v = 0.f;
  if (c < Cin) {
    float xv = x[((size_t)b * Cin + c) * L_ + t];
    if (useBN) {
      float s = g[c] * rsqrtf(var[c] + EPS_BN);
      v = xv * s + (bt[c] - mu[c] * s);
    } else v = xv;
  }
  y[idx] = (_Float16)v;
}

__global__ void bn_cast_f16(const _Float16* __restrict__ x, _Float16* __restrict__ y,
                            const float* __restrict__ g, const float* __restrict__ bt,
                            const float* __restrict__ mu, const float* __restrict__ var,
                            int C) {
  size_t idx = (size_t)blockIdx.x * blockDim.x + threadIdx.x;
  size_t total = (size_t)B_ * C * L_;
  if (idx >= total) return;
  int c = (int)((idx / L_) % C);
  float s = g[c] * rsqrtf(var[c] + EPS_BN);
  float v = (float)x[idx] * s + (bt[c] - mu[c] * s);
  y[idx] = (_Float16)v;
}

// conv weights f32 [Cout][Cin][taps] -> f16 A [Cout][taps][Cpad] (K-padded)
__global__ void cast_conv_w(const float* __restrict__ w, _Float16* __restrict__ A,
                            int Cout, int Cin, int Cpad, int taps) {
  int idx = blockIdx.x * blockDim.x + threadIdx.x;
  int total = Cout * taps * Cpad;
  if (idx >= total) return;
  int ci = idx % Cpad;
  int j  = (idx / Cpad) % taps;
  int co = idx / (Cpad * taps);
  A[idx] = (ci < Cin) ? (_Float16)w[((size_t)co * Cin + ci) * taps + j] : (_Float16)0.f;
}

// dense weights f32 [K][N] -> f16 [Kpad][N]
__global__ void cast_mat(const float* __restrict__ w, _Float16* __restrict__ o,
                         int K, int Kpad, int N) {
  int idx = blockIdx.x * blockDim.x + threadIdx.x;
  int total = Kpad * N;
  if (idx >= total) return;
  int n = idx % N;
  int k = idx / N;
  o[idx] = (k < K) ? (_Float16)w[(size_t)k * N + n] : (_Float16)0.f;
}

// static f32 [B*L][181] -> f16 [B*L][192]
__global__ void cast_static(const float* __restrict__ s, _Float16* __restrict__ o) {
  size_t idx = (size_t)blockIdx.x * blockDim.x + threadIdx.x;
  size_t total = (size_t)B_ * L_ * SPAD;
  if (idx >= total) return;
  int k = (int)(idx % SPAD);
  size_t m = idx / SPAD;
  o[idx] = (k < SDIM) ? (_Float16)s[m * SDIM + k] : (_Float16)0.f;
}

// hidden f16 [B][H][L] -> cat rows [B*L][640]
__global__ void build_cat(const _Float16* __restrict__ hid, _Float16* __restrict__ cat) {
  size_t idx = (size_t)blockIdx.x * blockDim.x + threadIdx.x;
  size_t total = (size_t)B_ * L_ * H_;
  if (idx >= total) return;
  int c = (int)(idx % H_);
  size_t m = idx / H_;
  int t = (int)(m % L_);
  int b = (int)(m / L_);
  _Float16 v = hid[((size_t)b * H_ + c) * L_ + t];
  cat[m * DIN + 128 + c] = (t > 0) ? v : (_Float16)0.f;
  cat[m * DIN + 384 + c] = v;
}

// ---------------- WMMA causal dilated conv (taps shifted GEMMs) ----------------
// Workgroup tile: 128 cout x 64 t; wave tile: 16 cout x 64 t (4 accumulators).
#define XW 80
#define XLD 34
__global__ __launch_bounds__(256) void conv_wmma(
    const _Float16* __restrict__ X, const _Float16* __restrict__ A,
    const float* __restrict__ bias, const _Float16* __restrict__ Res,
    _Float16* __restrict__ Y, int Cpad, int Cout, int taps, int dil, int mode) {
  __shared__ _Float16 sA[3 * 128 * 32];   // weight tiles, one per tap (24 KB), 64B rows
  __shared__ _Float16 sX[XW * XLD];       // transposed activation tile (5.3 KB)
  const int tid  = threadIdx.x;
  const int t0   = blockIdx.x * 64;
  const int co0  = blockIdx.y * 128;
  const int b    = blockIdx.z;
  const int wave = tid >> 5;
  const int lane = tid & 31;
  const int co_sub = wave * 16;
  const int kA = taps * Cpad;

  v8f acc[4];
#pragma unroll
  for (int nt = 0; nt < 4; ++nt) acc[nt] = (v8f){0.f,0.f,0.f,0.f,0.f,0.f,0.f,0.f};

  for (int ci0 = 0; ci0 < Cpad; ci0 += 32) {
    // ---- stage weight tiles for this K-slice ----
#if USE_ASYNC
    for (int cidx = tid; cidx < taps * 128 * 4; cidx += 256) {
      int j = cidx >> 9;            // /(128*4)
      int r = (cidx >> 2) & 127;
      int q = cidx & 3;             // 16-byte chunk within the 64B row
      const _Float16* g = &A[(size_t)(co0 + r) * kA + j * Cpad + ci0 + q * 8];
      async_cp16(g, &sA[(j * 128 + r) * 32 + q * 8]);
    }
#else
    for (int idx = tid; idx < taps * 128 * 32; idx += 256) {
      int j = idx >> 12;
      int r = (idx >> 5) & 127;
      int c = idx & 31;
      sA[idx] = A[(size_t)(co0 + r) * kA + j * Cpad + ci0 + c];
    }
#endif
    // ---- stage activation tile transposed: [t][ci] ----
    for (int idx = tid; idx < 32 * XW; idx += 256) {
      int r = idx / XW, c = idx % XW;   // r = ci, c = t-window col
      int t = t0 - 16 + c;
      _Float16 v = (_Float16)0.f;
      if (t >= 0 && t < L_) v = X[((size_t)b * Cpad + ci0 + r) * L_ + t];
      sX[c * XLD + r] = v;
    }
    // prefetch next channel tile (global_prefetch_b8)
    {
      int cn = (ci0 + 32 < Cpad) ? (ci0 + 32) : ci0;
      __builtin_prefetch(&X[((size_t)b * Cpad + cn + lane) * L_ + t0], 0, 1);
    }
#if USE_ASYNC
    wait_async0();
#endif
    __syncthreads();
    for (int j = 0; j < taps; ++j) {
      v16h a = load_a_frag(&sA[(j * 128 + co_sub) * 32], 32);
      const int off = 16 - (taps - 1 - j) * dil;   // window row of t0 for this tap
#pragma unroll
      for (int nt = 0; nt < 4; ++nt) {
        v16h bb = load_b_fragT(&sX[(off + nt * 16) * XLD], XLD);
        acc[nt] = __builtin_amdgcn_wmma_f32_16x16x32_f16(false, a, false, bb,
                                                         (short)0, acc[nt], false, false);
      }
    }
    __syncthreads();
  }

  const int nb = lane & 15, hi = lane >> 4;
#pragma unroll
  for (int r = 0; r < 8; ++r) {
    int m = r + hi * 8;
    int co = co0 + co_sub + m;
    float bsv = bias[co];
    size_t rowBase = ((size_t)b * Cout + co) * L_;
#pragma unroll
    for (int nt = 0; nt < 4; ++nt) {
      int t = t0 + nt * 16 + nb;
      float v = acc[nt][r] + bsv;
      if (mode >= 1) v = selu_f(v);
      if (mode == 2) v = selu_f(v + (float)Res[rowBase + t]);
      Y[rowBase + t] = (_Float16)v;
    }
  }
}

// ---------------- WMMA dense GEMM: act[M][K] x W[K][N] (+bias, selu) -> C f16 ----------------
#define ALD 40
__global__ __launch_bounds__(256) void gemm_wmma(
    const _Float16* __restrict__ Aact, const _Float16* __restrict__ Wt,
    const float* __restrict__ bias, _Float16* __restrict__ C,
    int K, int N, int ldc, int coff, int act) {
  __shared__ _Float16 sAct[128 * ALD];  // row-major [m][k], 80B row stride (16B aligned)
  __shared__ _Float16 sWT[64 * 34];     // transposed weights [n][k]
  const int tid = threadIdx.x;
  const int m0 = blockIdx.x * 128;
  const int n0 = blockIdx.y * 64;
  const int wave = tid >> 5, lane = tid & 31;

  v8f acc[4];
#pragma unroll
  for (int nt = 0; nt < 4; ++nt) acc[nt] = (v8f){0.f,0.f,0.f,0.f,0.f,0.f,0.f,0.f};

  for (int k0 = 0; k0 < K; k0 += 32) {
#if USE_ASYNC
    for (int cidx = tid; cidx < 128 * 4; cidx += 256) {
      int r = cidx >> 2;
      int q = cidx & 3;
      async_cp16(&Aact[(size_t)(m0 + r) * K + k0 + q * 8], &sAct[r * ALD + q * 8]);
    }
#else
    for (int idx = tid; idx < 128 * 32; idx += 256) {
      int r = idx >> 5, c = idx & 31;
      sAct[r * ALD + c] = Aact[(size_t)(m0 + r) * K + k0 + c];
    }
#endif
    for (int idx = tid; idx < 32 * 64; idx += 256) {
      int r = idx >> 6, c = idx & 63;   // r = k, c = n (coalesced along n)
      sWT[c * 34 + r] = Wt[(size_t)(k0 + r) * N + n0 + c];
    }
#if USE_ASYNC
    wait_async0();
#endif
    __syncthreads();
    v16h a = load_a_frag(&sAct[(wave * 16) * ALD], ALD);
#pragma unroll
    for (int nt = 0; nt < 4; ++nt) {
      v16h bb = load_b_fragT(&sWT[(nt * 16) * 34], 34);
      acc[nt] = __builtin_amdgcn_wmma_f32_16x16x32_f16(false, a, false, bb,
                                                       (short)0, acc[nt], false, false);
    }
    __syncthreads();
  }

  const int nb = lane & 15, hi = lane >> 4;
#pragma unroll
  for (int r = 0; r < 8; ++r) {
    int m = m0 + wave * 16 + r + hi * 8;
#pragma unroll
    for (int nt = 0; nt < 4; ++nt) {
      int n = n0 + nt * 16 + nb;
      float v = acc[nt][r] + bias[n];
      if (act) v = selu_f(v);
      C[(size_t)m * ldc + coff + n] = (_Float16)v;
    }
  }
}

// ---------------- final GEMV: out[m] = h[m][:] . Wo + bo ----------------
__global__ void gemv_out(const _Float16* __restrict__ Hf, const float* __restrict__ Wo,
                         const float* __restrict__ bo, float* __restrict__ out,
                         int M, int K) {
  int wid = (int)(((size_t)blockIdx.x * blockDim.x + threadIdx.x) >> 5);
  int lane = threadIdx.x & 31;
  if (wid >= M) return;
  float acc = 0.f;
  for (int k = lane; k < K; k += 32)
    acc += (float)Hf[(size_t)wid * K + k] * Wo[k];
#pragma unroll
  for (int off = 16; off; off >>= 1) acc += __shfl_xor(acc, off, 32);
  if (lane == 0) out[wid] = acc + bo[0];
}

// ---------------- host ----------------
extern "C" void kernel_launch(void* const* d_in, const int* in_sizes, int n_in,
                              void* d_out, int out_size, void* d_ws, size_t ws_size,
                              hipStream_t stream) {
  (void)n_in; (void)out_size; (void)ws_size;
  auto F = [&](int i) { return (const float*)d_in[i]; };

  const float *p_static, *p_dyn, *Ws, *bs, *Wf, *bf, *Wo, *bo;
  struct Blk { const float *g1,*b1v,*m1,*v1,*w1,*bi1,*g2,*b2v,*m2,*v2,*w2,*bi2,*dw,*db; } blk[4];

  if (in_sizes[0] == B_ * L_ * SDIM) {
    // dict-insertion-order flattening
    p_static = F(0); p_dyn = F(1);
    int i = 2;
    for (int k = 0; k < 4; ++k) {
      blk[k].g1 = F(i++); blk[k].b1v = F(i++); blk[k].m1 = F(i++); blk[k].v1 = F(i++);
      blk[k].w1 = F(i++); blk[k].bi1 = F(i++);
      blk[k].g2 = F(i++); blk[k].b2v = F(i++); blk[k].m2 = F(i++); blk[k].v2 = F(i++);
      blk[k].w2 = F(i++); blk[k].bi2 = F(i++);
      if (k == 0) { blk[k].dw = F(i++); blk[k].db = F(i++); }
      else { blk[k].dw = nullptr; blk[k].db = nullptr; }
    }
    i += 4; // skip attn (reference masks it out of the output)
    Ws = F(i++); bs = F(i++); Wf = F(i++); bf = F(i++); Wo = F(i++); bo = F(i++);
  } else {
    // jax tree (sorted-key) flattening: attn, dynamic, head, static, tcn
    p_dyn = F(4);
    Wf = F(5); Wo = F(6); Ws = F(7); bf = F(8); bo = F(9); bs = F(10);
    p_static = F(11);
    int i = 12;
    for (int k = 0; k < 4; ++k) {
      blk[k].bi1 = F(i++); blk[k].bi2 = F(i++);
      blk[k].b1v = F(i++); blk[k].g1 = F(i++); blk[k].m1 = F(i++); blk[k].v1 = F(i++);
      blk[k].b2v = F(i++); blk[k].g2 = F(i++); blk[k].m2 = F(i++); blk[k].v2 = F(i++);
      if (k == 0) { blk[k].db = F(i++); blk[k].dw = F(i++); }
      else { blk[k].dw = nullptr; blk[k].db = nullptr; }
      blk[k].w1 = F(i++); blk[k].w2 = F(i++);
    }
  }

  // ---- workspace carve-up ----
  char* p = (char*)d_ws;
  auto alloc = [&](size_t bytes) {
    void* r = (void*)p;
    p += (bytes + 255) & ~(size_t)255;
    return r;
  };
  const size_t M = (size_t)B_ * L_;
  _Float16* bufX   = (_Float16*)alloc((size_t)B_ * 256 * L_ * 2);
  _Float16* bufBN  = (_Float16*)alloc((size_t)B_ * 256 * L_ * 2);
  _Float16* bufT   = (_Float16*)alloc((size_t)B_ * 256 * L_ * 2);
  _Float16* bufR   = (_Float16*)alloc((size_t)B_ * 256 * L_ * 2);
  _Float16* catBuf = (_Float16*)alloc(M * DIN * 2);
  _Float16* hBuf   = (_Float16*)alloc(M * FCL * 2);
  _Float16* staticH= (_Float16*)alloc(M * SPAD * 2);
  _Float16* wA[4], *wB[4];
  wA[0] = (_Float16*)alloc((size_t)256 * 3 * 96 * 2);
  for (int k = 1; k < 4; ++k) wA[k] = (_Float16*)alloc((size_t)256 * 3 * 256 * 2);
  for (int k = 0; k < 4; ++k) wB[k] = (_Float16*)alloc((size_t)256 * 3 * 256 * 2);
  _Float16* wD  = (_Float16*)alloc((size_t)256 * 96 * 2);
  _Float16* WsH = (_Float16*)alloc((size_t)SPAD * EMB * 2);
  _Float16* WfH = (_Float16*)alloc((size_t)DIN * FCL * 2);

  auto gridFor = [](size_t n) { return (unsigned)((n + 255) / 256); };

  // ---- weight casts ----
  cast_conv_w<<<gridFor(256 * 3 * 96), 256, 0, stream>>>(blk[0].w1, wA[0], 256, 74, 96, 3);
  for (int k = 1; k < 4; ++k)
    cast_conv_w<<<gridFor(256 * 3 * 256), 256, 0, stream>>>(blk[k].w1, wA[k], 256, 256, 256, 3);
  for (int k = 0; k < 4; ++k)
    cast_conv_w<<<gridFor(256 * 3 * 256), 256, 0, stream>>>(blk[k].w2, wB[k], 256, 256, 256, 3);
  cast_conv_w<<<gridFor(256 * 96), 256, 0, stream>>>(blk[0].dw, wD, 256, 74, 96, 1);
  cast_mat<<<gridFor((size_t)SPAD * EMB), 256, 0, stream>>>(Ws, WsH, SDIM, SPAD, EMB);
  cast_mat<<<gridFor((size_t)DIN * FCL), 256, 0, stream>>>(Wf, WfH, DIN, DIN, FCL);
  cast_static<<<gridFor(M * SPAD), 256, 0, stream>>>(p_static, staticH);

  // ---- TCN block 0 (cin 74 -> pad 96, dil 1, downsample residual) ----
  bn_cast_f32<<<gridFor((size_t)B_ * 96 * L_), 256, 0, stream>>>(
      p_dyn, bufBN, blk[0].g1, blk[0].b1v, blk[0].m1, blk[0].v1, 74, 96, 1);
  bn_cast_f32<<<gridFor((size_t)B_ * 96 * L_), 256, 0, stream>>>(
      p_dyn, bufX, nullptr, nullptr, nullptr, nullptr, 74, 96, 0);
  {
    dim3 g(L_ / 64, 256 / 128, B_);
    conv_wmma<<<g, 256, 0, stream>>>(bufX, wD, blk[0].db, nullptr, bufR, 96, 256, 1, 1, 0);
    conv_wmma<<<g, 256, 0, stream>>>(bufBN, wA[0], blk[0].bi1, nullptr, bufT, 96, 256, 3, 1, 1);
  }
  bn_cast_f16<<<gridFor((size_t)B_ * 256 * L_), 256, 0, stream>>>(
      bufT, bufBN, blk[0].g2, blk[0].b2v, blk[0].m2, blk[0].v2, 256);
  {
    dim3 g(L_ / 64, 256 / 128, B_);
    conv_wmma<<<g, 256, 0, stream>>>(bufBN, wB[0], blk[0].bi2, bufR, bufX, 256, 256, 3, 1, 2);
  }

  // ---- TCN blocks 1..3 (identity residual, dil 2/4/8) ----
  for (int k = 1; k < 4; ++k) {
    int dil = 1 << k;
    bn_cast_f16<<<gridFor((size_t)B_ * 256 * L_), 256, 0, stream>>>(
        bufX, bufBN, blk[k].g1, blk[k].b1v, blk[k].m1, blk[k].v1, 256);
    dim3 g(L_ / 64, 256 / 128, B_);
    conv_wmma<<<g, 256, 0, stream>>>(bufBN, wA[k], blk[k].bi1, nullptr, bufT, 256, 256, 3, dil, 1);
    bn_cast_f16<<<gridFor((size_t)B_ * 256 * L_), 256, 0, stream>>>(
        bufT, bufBN, blk[k].g2, blk[k].b2v, blk[k].m2, blk[k].v2, 256);
    conv_wmma<<<g, 256, 0, stream>>>(bufBN, wB[k], blk[k].bi2, bufX, bufX, 256, 256, 3, dil, 2);
  }

  // ---- head ----
  build_cat<<<gridFor(M * H_), 256, 0, stream>>>(bufX, catBuf);
  {
    dim3 g((unsigned)(M / 128), EMB / 64);
    gemm_wmma<<<g, 256, 0, stream>>>(staticH, WsH, bs, catBuf, SPAD, EMB, DIN, 0, 1);
  }
  {
    dim3 g((unsigned)(M / 128), FCL / 64);
    gemm_wmma<<<g, 256, 0, stream>>>(catBuf, WfH, bf, hBuf, DIN, FCL, FCL, 0, 1);
  }
  gemv_out<<<gridFor(M * 32), 256, 0, stream>>>(hBuf, Wo, bo, (float*)d_out, (int)M, FCL);
}
